// Net_SLSTM_Extractor_34127810134433
// MI455X (gfx1250) — compile-verified, matrix operand order checked by reading, not run on hardware
//
#include <hip/hip_runtime.h>

// ---------------------------------------------------------------------------
// 2-layer SLSTM, persistent-recurrent kernel for gfx1250 (MI455X).
// B=512 -> 32 workgroups x 16 batch rows; recurrent state lives on-WGP for
// all T=1000 steps. Matmuls: v_wmma_f32_16x16x32_f16, f32 accumulate.
// Weights: f16 B-fragments packed in d_ws, streamed from L2 every step with a
// laundered *offset* (keeps global addrspace -> global_load_b128, not flat)
// and a 2-stage software pipeline so loads overlap the WMMAs.
// Shared recurrent state (mem1/spk1/mem2) and the staged x tile live in LDS
// as f16 in A-fragment order -> every A operand is two ds_load_b128.
// Lane-private state (syn1/syn2, mem copies, running mean) stays in VGPRs.
// ---------------------------------------------------------------------------

#define T_STEPS 1000
#define BATCH   512
#define IN_DIM  14
#define HID     128
#define BT      16              // batch tile (WMMA M)
#define NWGS    (BATCH / BT)    // 32 workgroups

// fragment table: each fragment = one [K=32 x N=16] f16 B-operand, 512 halves
#define NFRAG_HALFS 512
#define F_WIH1 0                //  32 frags (K-chunk 0, zero-padded past IN_DIM)
#define F_WHH1 32               // 128 frags (32 n-tiles x 4 K-chunks)
#define F_WIH2 160              // 128 frags
#define F_WHH2 288              // 128 frags
#define NFRAGS 416

#define WS_BIAS_BYTES ((size_t)NFRAGS * NFRAG_HALFS * 2)   // f16 frags, then f32 biases

#define OUT_M1 ((size_t)BATCH * HID)                         // 65536
#define OUT_M2 (OUT_M1 + (size_t)T_STEPS * BATCH * HID)      // 65601536

// f16 state arrays in LDS, stored directly in 16-bit A-operand order:
// halfIndex(c, kgrp, M, h) = ((c*2 + kgrp)*16 + M)*16 + h
#define ST_HALFS (4 * 2 * 16 * 16)   // 2048 halves = 4 KB per array
#define XH_HALFS (2 * 16 * 16)       // x tile: K-chunk 0 only, 1 KB per buffer

typedef __attribute__((ext_vector_type(16))) _Float16 v16h;
typedef __attribute__((ext_vector_type(8)))  float    v8f;

// 16-bit A/B-operand half->K mapping (ISA 7.12.2, 16-bit A 16x32 layout):
// lanes 0-15 hold K {0..7, 16..23}; lanes 16-31 hold K {8..15, 24..31}.
__device__ __forceinline__ int kmap(int lane, int h) {
  int v = h >> 1, pos = h & 1;
  int k = (v < 4) ? (2 * v + pos) : (16 + 2 * (v - 4) + pos);
  return k + 8 * (lane >> 4);
}
// closed-form inverse for a K value within a chunk: which (kgrp, half) slot
__device__ __forceinline__ void inv_kmap(int kl, int& kgrp, int& h) {
  kgrp = (kl >> 3) & 1;
  int base = kl - 8 * kgrp;                  // {0..7} u {16..23}
  h = (base < 8) ? base : base - 8;
}

__device__ __forceinline__ float sig_f(float x)  { return 1.0f / (1.0f + __expf(-x)); }
__device__ __forceinline__ float tanh_f(float x) { return 1.0f - 2.0f / (__expf(2.0f * x) + 1.0f); }

__device__ __forceinline__ v16h load_frag_g(const _Float16* p, int f, int lane) {
  return *(const v16h*)(p + (size_t)f * NFRAG_HALFS + lane * 16);
}
__device__ __forceinline__ v16h load_frag_lds(const _Float16* s, int c, int lane) {
  return *(const v16h*)(s + ((c * 2 + (lane >> 4)) * 16 + (lane & 15)) * 16);
}

// ---------------------------------------------------------------------------
// Prep: pack weights into f16 B-fragments. grid = NFRAGS x 512 threads.
// ---------------------------------------------------------------------------
__global__ void slstm_pack_weights(const float* __restrict__ Wih1,
                                   const float* __restrict__ Whh1,
                                   const float* __restrict__ Wih2,
                                   const float* __restrict__ Whh2,
                                   _Float16* __restrict__ wfr) {
  const int f   = blockIdx.x;
  const int tid = threadIdx.x;       // 0..511
  const int l   = tid >> 4;          // lane 0..31
  const int h   = tid & 15;          // half 0..15
  const int kl  = kmap(l, h);        // 0..31 within K-chunk
  const int nl  = l & 15;            // N within tile

  float val;
  if (f < F_WHH1) {                                   // Wih1 [512][14], K-chunk 0
    int n = f * 16 + nl;
    val = (kl < IN_DIM) ? Wih1[n * IN_DIM + kl] : 0.0f;
  } else if (f < F_WIH2) {                            // Whh1 [512][128]
    int i = f - F_WHH1, nt = i >> 2, c = i & 3;
    val = Whh1[(nt * 16 + nl) * HID + c * 32 + kl];
  } else if (f < F_WHH2) {                            // Wih2 [512][128]
    int i = f - F_WIH2, nt = i >> 2, c = i & 3;
    val = Wih2[(nt * 16 + nl) * HID + c * 32 + kl];
  } else {                                            // Whh2 [512][128]
    int i = f - F_WHH2, nt = i >> 2, c = i & 3;
    val = Whh2[(nt * 16 + nl) * HID + c * 32 + kl];
  }
  wfr[(size_t)f * NFRAG_HALFS + l * 16 + h] = (_Float16)val;
}

__global__ void slstm_pack_bias(const float* __restrict__ bih1,
                                const float* __restrict__ bhh1,
                                const float* __restrict__ bih2,
                                const float* __restrict__ bhh2,
                                float* __restrict__ bias) {
  int i = blockIdx.x * blockDim.x + threadIdx.x;     // 0..1023
  if (i < 512)       bias[i] = bih1[i] + bhh1[i];
  else if (i < 1024) bias[i] = bih2[i - 512] + bhh2[i - 512];
}

// ---------------------------------------------------------------------------
// Main persistent kernel: 32 blocks x 256 threads (8 waves).
// Wave w owns gate columns {w*16 + g*128 : g=0..3}: each lane holds i/f/g/o
// for the same (row, h) across its four accumulators.
// ---------------------------------------------------------------------------
__global__ __launch_bounds__(256)
void slstm_persistent(const float* __restrict__ x,
                      const _Float16* __restrict__ wfr,
                      const float* __restrict__ bias,
                      const float* __restrict__ thr1p,
                      const float* __restrict__ thr2p,
                      float* __restrict__ out) {
  __shared__ _Float16 s_m1h[ST_HALFS];      // mem1, f16, A-fragment order
  __shared__ _Float16 s_s1h[ST_HALFS];      // spk1
  __shared__ _Float16 s_m2h[ST_HALFS];      // mem2
  __shared__ _Float16 s_xh[2][XH_HALFS];    // staged x tile (double buffer)

  const int tid  = threadIdx.x;
  const int lane = tid & 31;
  const int wave = tid >> 5;                    // 0..7
  const int b0   = blockIdx.x * BT;

  const float thr1 = thr1p[0];
  const float thr2 = thr2p[0];

  for (int i = tid; i < ST_HALFS; i += 256) {
    s_m1h[i] = (_Float16)0.0f;
    s_s1h[i] = (_Float16)0.0f;
    s_m2h[i] = (_Float16)0.0f;
  }
  for (int i = tid; i < 2 * XH_HALFS; i += 256)   // zero incl. padding slots
    s_xh[0][i] = (_Float16)0.0f;

  // x-tile staging slot for this thread (thread -> one element of 16x14 tile)
  const bool xStager = (tid < BT * IN_DIM);
  int xM = 0, xSlot = 0;
  if (xStager) {
    xM = tid / IN_DIM;
    int xk = tid % IN_DIM, kg, hh;
    inv_kmap(xk, kg, hh);
    xSlot = (kg * 16 + xM) * 16 + hh;
  }
  // stage x for t=0 into buffer 0
  if (xStager)
    s_xh[0][xSlot] = (_Float16)x[(size_t)(b0 + xM) * IN_DIM + (tid % IN_DIM)];
  __syncthreads();

  const int hC    = wave * 16 + (lane & 15);    // hidden column in elementwise
  const int Mbase = 8 * (lane >> 4);            // C/D layout row base

  // LDS store slot for this lane's hidden column hC (A-layout inverse)
  int stKgrp, stH;
  inv_kmap(hC & 31, stKgrp, stH);
  const int stOff = (((hC >> 5) * 2 + stKgrp) * 16) * 16 + stH;   // + Mr*16

  // lane-private recurrent state (each (row,h) cell owned by one lane)
  float syn1r[8], mem1r[8], syn2r[8], mem2r[8], fsum[8];
#pragma unroll
  for (int r = 0; r < 8; ++r) {
    syn1r[r] = 0.f; mem1r[r] = 0.f; syn2r[r] = 0.f; mem2r[r] = 0.f; fsum[r] = 0.f;
  }

  float b1v[4], b2v[4];
#pragma unroll
  for (int g = 0; g < 4; ++g) {
    b1v[g] = bias[g * HID + hC];
    b2v[g] = bias[512 + g * HID + hC];
  }

#pragma unroll 1
  for (int t = 0; t < T_STEPS; ++t) {
    // Launder a zero OFFSET (not the pointer!) so LICM cannot hoist the
    // loop-invariant weight loads (-> ~550 VGPRs + scratch spills), while the
    // pointer keeps its global address space (global_load_b128, LOADcnt only).
    unsigned long long woff = 0;
    asm volatile("" : "+s"(woff));
    const _Float16* wp = wfr + woff;

    // ---- phase 1: A fragments for mem1 + stage next x tile
    v16h am1[4];
#pragma unroll
    for (int c = 0; c < 4; ++c) am1[c] = load_frag_lds(s_m1h, c, lane);
    if (xStager && (t + 1 < T_STEPS)) {
      const float* xn = x + (size_t)(t + 1) * (BATCH * IN_DIM) + (size_t)(b0 + xM) * IN_DIM + (tid % IN_DIM);
      s_xh[(t + 1) & 1][xSlot] = (_Float16)xn[0];
      if (t + 2 < T_STEPS)
        __builtin_prefetch(xn + BATCH * IN_DIM, 0, 1);   // warm L2 for t+2
    }
    __syncthreads();                            // mem1 reads done; x(t) ready

    // ---- phase 2: layer-1 WMMA, 2-stage pipelined B-fragment stream
    v8f acc[4];
#pragma unroll
    for (int g = 0; g < 4; ++g) {
      v8f a;
#pragma unroll
      for (int r = 0; r < 8; ++r) a[r] = b1v[g];
      acc[g] = a;
    }
    {
      const v16h ax = load_frag_lds(s_xh[t & 1], 0, lane);
      v16h curB[4], nxtB[4];
#pragma unroll
      for (int g = 0; g < 4; ++g) curB[g] = load_frag_g(wp, F_WIH1 + (g * 8 + wave), lane);
#pragma unroll
      for (int s = 0; s < 5; ++s) {             // stage 0: x, stages 1..4: mem1
        if (s < 4) {
#pragma unroll
          for (int g = 0; g < 4; ++g)
            nxtB[g] = load_frag_g(wp, F_WHH1 + (g * 8 + wave) * 4 + s, lane);
        }
        const v16h As = (s == 0) ? ax : am1[s - 1];
#pragma unroll
        for (int g = 0; g < 4; ++g)
          acc[g] = __builtin_amdgcn_wmma_f32_16x16x32_f16(false, As, false, curB[g], (short)0, acc[g], false, false);
        if (s < 4) {
#pragma unroll
          for (int g = 0; g < 4; ++g) curB[g] = nxtB[g];
        }
      }
    }

    // ---- phase 3: layer-1 elementwise (registers); publish f16 mem1/spk1
#pragma unroll
    for (int r = 0; r < 8; ++r) {
      const int Mr = r + Mbase;
      float gi = acc[0][r], gf = acc[1][r], gg = acc[2][r], go = acc[3][r];
      float cc = sig_f(gf) * syn1r[r] + sig_f(gi) * tanh_f(gg);
      float hh = sig_f(go) * tanh_f(cc);
      float mn = hh - ((mem1r[r] > thr1) ? thr1 : 0.0f);   // reset from PREVIOUS mem
      syn1r[r] = cc;
      mem1r[r] = mn;
      s_m1h[stOff + Mr * 16] = (_Float16)mn;
      s_s1h[stOff + Mr * 16] = (_Float16)((mn > thr1) ? 1.0f : 0.0f);
      out[OUT_M1 + (size_t)t * (BATCH * HID) + (size_t)(b0 + Mr) * HID + hC] = mn;
    }
    __syncthreads();                            // spk1/mem1 published

    // ---- phase 4: layer-2 A fragments (spk1, mem2 from LDS f16)
    v16h as2[4], am2[4];
#pragma unroll
    for (int c = 0; c < 4; ++c) {
      as2[c] = load_frag_lds(s_s1h, c, lane);
      am2[c] = load_frag_lds(s_m2h, c, lane);
    }
    __syncthreads();                            // mem2 reads done before rewrite

    // ---- phase 5: layer-2 WMMA (pipelined) + elementwise
#pragma unroll
    for (int g = 0; g < 4; ++g) {
      v8f a;
#pragma unroll
      for (int r = 0; r < 8; ++r) a[r] = b2v[g];
      acc[g] = a;
    }
    {
      v16h curB[4], nxtB[4];
#pragma unroll
      for (int g = 0; g < 4; ++g) curB[g] = load_frag_g(wp, F_WIH2 + (g * 8 + wave) * 4, lane);
#pragma unroll
      for (int s = 0; s < 8; ++s) {             // stages 0..3: spk1, 4..7: mem2
        if (s < 7) {
          const int fs = s + 1;
          const int fb = (fs < 4) ? F_WIH2 : F_WHH2;
          const int cc = (fs < 4) ? fs : fs - 4;
#pragma unroll
          for (int g = 0; g < 4; ++g)
            nxtB[g] = load_frag_g(wp, fb + (g * 8 + wave) * 4 + cc, lane);
        }
        const v16h As = (s < 4) ? as2[s] : am2[s - 4];
#pragma unroll
        for (int g = 0; g < 4; ++g)
          acc[g] = __builtin_amdgcn_wmma_f32_16x16x32_f16(false, As, false, curB[g], (short)0, acc[g], false, false);
        if (s < 7) {
#pragma unroll
          for (int g = 0; g < 4; ++g) curB[g] = nxtB[g];
        }
      }
    }
#pragma unroll
    for (int r = 0; r < 8; ++r) {
      const int Mr = r + Mbase;
      float gi = acc[0][r], gf = acc[1][r], gg = acc[2][r], go = acc[3][r];
      float cc = sig_f(gf) * syn2r[r] + sig_f(gi) * tanh_f(gg);
      float hh = sig_f(go) * tanh_f(cc);
      float mn = hh - ((mem2r[r] > thr2) ? thr2 : 0.0f);
      syn2r[r] = cc;
      mem2r[r] = mn;
      s_m2h[stOff + Mr * 16] = (_Float16)mn;
      fsum[r] += mn;
      out[OUT_M2 + (size_t)t * (BATCH * HID) + (size_t)(b0 + Mr) * HID + hC] = mn;
    }
    // next iteration's phase-1/phase-3 barriers provide the remaining ordering
  }

  // finalMem = mean over time of mem2
#pragma unroll
  for (int r = 0; r < 8; ++r) {
    const int Mr = r + Mbase;
    out[(size_t)(b0 + Mr) * HID + hC] = fsum[r] * (1.0f / (float)T_STEPS);
  }
}

// ---------------------------------------------------------------------------
extern "C" void kernel_launch(void* const* d_in, const int* in_sizes, int n_in,
                              void* d_out, int out_size, void* d_ws, size_t ws_size,
                              hipStream_t stream) {
  const float* x    = (const float*)d_in[0];
  const float* Wih1 = (const float*)d_in[1];
  const float* Whh1 = (const float*)d_in[2];
  const float* bih1 = (const float*)d_in[3];
  const float* bhh1 = (const float*)d_in[4];
  const float* thr1 = (const float*)d_in[5];
  const float* Wih2 = (const float*)d_in[6];
  const float* Whh2 = (const float*)d_in[7];
  const float* bih2 = (const float*)d_in[8];
  const float* bhh2 = (const float*)d_in[9];
  const float* thr2 = (const float*)d_in[10];

  _Float16* wfr  = (_Float16*)d_ws;
  float*    bias = (float*)((char*)d_ws + WS_BIAS_BYTES);
  float*    out  = (float*)d_out;

  slstm_pack_weights<<<NFRAGS, 512, 0, stream>>>(Wih1, Whh1, Wih2, Whh2, wfr);
  slstm_pack_bias<<<2, 512, 0, stream>>>(bih1, bhh1, bih2, bhh2, bias);
  slstm_persistent<<<NWGS, 256, 0, stream>>>(x, wfr, bias, thr1, thr2, out);
}
